// Network_8057358647751
// MI455X (gfx1250) — compile-verified, matrix operand order checked by reading, not run on hardware
//
#include <hip/hip_runtime.h>
#include <cstddef>

// ---------------- constants ----------------
constexpr int L    = 10000;
constexpr int DIN  = 1024;
constexpr int DM   = 256;
constexpr int DI   = 512;
constexpr int CH   = 100;   // scan chunks
constexpr int SC   = 100;   // steps per chunk (CH*SC == L)

typedef _Float16 half_t;
typedef __attribute__((ext_vector_type(16))) _Float16 v16h;
typedef __attribute__((ext_vector_type(8)))  _Float16 v8h;
typedef __attribute__((ext_vector_type(8)))  float    v8f;

// ---------------- workspace layout (bytes) ----------------
constexpr size_t OF_H     = 0;                       // L*DM*4
constexpr size_t OF_H0    = 10240000;                // L*DM*4
constexpr size_t OF_XPERM = 20480000;                // L*DI*4 (aliased by DELTA)
constexpr size_t OF_Z     = 40960000;                // L*DI*4
constexpr size_t OF_XC    = 61440000;                // L*DI*4
constexpr size_t OF_XDBL  = 81920000;                // L*64*4
constexpr size_t OF_TY16  = 84480000;                // t16 (L*DM*2) then y16 (L*DI*2)
constexpr size_t OF_SUMD  = 104960000;               // CH*DI*4
constexpr size_t OF_SACC  = 105164800;               // CH*8192*4
constexpr size_t OF_SINIT = 108441600;               // CH*8192*4
constexpr size_t OF_RED   = 111718400;
constexpr size_t OF_FEAT  = 111718464;
constexpr size_t OF_W16   = 111720448;               // f16 weight staging (~2.2MB)

// f16 weight segment sizes (elements)
constexpr size_t WN_FC1 = (size_t)DM * DIN;          // 262144
constexpr size_t WN_IN  = (size_t)2 * DI * DM;       // 262144
constexpr size_t WN_XP  = (size_t)48 * DI;           // 24576
constexpr size_t WN_DT  = (size_t)DI * 16;           // 8192
constexpr size_t WN_OUT = (size_t)DM * DI;           // 131072
constexpr size_t WN_BLK = WN_IN + WN_XP + WN_DT + WN_OUT;

// ---------------- WMMA fragment loaders ----------------
// 16-bit operand layout (CDNA5 ISA 7.12.2): lane = {kgrp(1b), row(4b)};
// frag element e -> K = kgrp*8 + ((e&8)<<1) + (e&7).
__device__ __forceinline__ v16h load_frag(const float* __restrict__ row, int k0, int kg) {
  const float* p = row + k0 + kg * 8;
  float4 a = *reinterpret_cast<const float4*>(p);
  float4 b = *reinterpret_cast<const float4*>(p + 4);
  float4 c = *reinterpret_cast<const float4*>(p + 16);
  float4 d = *reinterpret_cast<const float4*>(p + 20);
  v16h f;
  f[0]=(half_t)a.x;  f[1]=(half_t)a.y;  f[2]=(half_t)a.z;  f[3]=(half_t)a.w;
  f[4]=(half_t)b.x;  f[5]=(half_t)b.y;  f[6]=(half_t)b.z;  f[7]=(half_t)b.w;
  f[8]=(half_t)c.x;  f[9]=(half_t)c.y;  f[10]=(half_t)c.z; f[11]=(half_t)c.w;
  f[12]=(half_t)d.x; f[13]=(half_t)d.y; f[14]=(half_t)d.z; f[15]=(half_t)d.w;
  return f;
}

__device__ __forceinline__ v16h load_frag(const half_t* __restrict__ row, int k0, int kg) {
  const v8h* p0 = reinterpret_cast<const v8h*>(row + k0 + kg * 8);
  const v8h* p1 = reinterpret_cast<const v8h*>(row + k0 + kg * 8 + 16);
  v8h a = *p0, b = *p1;
  v16h f;
#pragma unroll
  for (int e = 0; e < 8; ++e) { f[e] = a[e]; f[e + 8] = b[e]; }
  return f;
}

template <class T>
__device__ __forceinline__ v16h load_frag_g(const T* __restrict__ row, int k0, int kg, int K) {
  v16h f;
#pragma unroll
  for (int e = 0; e < 16; ++e) {
    int k = k0 + kg * 8 + ((e & 8) << 1) + (e & 7);
    f[e] = (k < K) ? (half_t)row[k] : (half_t)0.f;
  }
  return f;
}

// ---------------- GEMM epilogues ----------------
struct EpiFC1 {            // + bias, exact GELU -> h (L,256)
  const float* bias; float* C;
  __device__ void operator()(int m, int n, float v) const {
    float x = v + bias[n];
    C[(size_t)m * DM + n] = 0.5f * x * (1.f + erff(x * 0.70710678118654752f));
  }
};
struct EpiInProj {         // split x/z; scatter x into permuted domain
  float* xperm; float* z;
  __device__ void operator()(int m, int n, float v) const {
    if (n < DI) {
      int p = (m % 10) * 1000 + m / 10;        // inv[m]
      xperm[(size_t)p * DI + n] = v;
    } else {
      z[(size_t)m * DI + (n - DI)] = v;
    }
  }
};
struct EpiXdbl {           // plain store, ld=64 (cols 0..47 used)
  float* C;
  __device__ void operator()(int m, int n, float v) const { C[(size_t)m * 64 + n] = v; }
};
struct EpiDT {             // + dtproj_b, softplus -> delta (L,512)
  const float* bias; float* delta;
  __device__ void operator()(int m, int n, float v) const {
    float x = v + bias[n];
    delta[(size_t)m * DI + n] = (x > 20.f) ? x : log1pf(__expf(x));
  }
};
struct EpiOutB {           // mode 0: h0 = h + out1 ; mode 1: hf = 0.5*(h0 + h[m] + out3) at q=L-1-m
  const float* h; float* h0; int mode;
  __device__ void operator()(int m, int n, float v) const {
    if (mode == 0) {
      h0[(size_t)m * DM + n] = h[(size_t)m * DM + n] + v;
    } else {
      int q = (L - 1) - m;
      size_t iq = (size_t)q * DM + n;
      h0[iq] = 0.5f * (h0[iq] + h[(size_t)m * DM + n] + v);
    }
  }
};

// ---------------- WMMA GEMM: C(MxN) = A(MxK,row) * W(NxK,row)^T ----------------
// A: f32 or f16 row-major; B: pre-converted f16 (W[n][k]).
// block = 128 threads (4 waves); each wave: one 16-row tile x 64 columns.
template <class TA, bool KG, bool NG, class Epi>
__global__ void __launch_bounds__(128)
gemm_wmma(const TA* __restrict__ A, int lda,
          const half_t* __restrict__ B, int ldb,
          int M, int N, int K, Epi epi) {
  const int lane = threadIdx.x & 31;
  const int wave = threadIdx.x >> 5;
  const int mtile = blockIdx.x * 4 + wave;
  if (mtile * 16 >= M) return;               // uniform per wave
  const int kg   = lane >> 4;
  const int lidx = lane & 15;
  const TA* arow = A + (size_t)(mtile * 16 + lidx) * lda;
  const int n0 = blockIdx.y * 64;

  const half_t* brow[4];
  bool bval[4];
#pragma unroll
  for (int t = 0; t < 4; ++t) {
    int n = n0 + t * 16 + lidx;
    bval[t] = (!NG) || (n < N);
    brow[t] = B + (size_t)(bval[t] ? n : 0) * ldb;
  }

  v8f acc[4];
#pragma unroll
  for (int t = 0; t < 4; ++t)
#pragma unroll
    for (int v = 0; v < 8; ++v) acc[t][v] = 0.f;

  for (int k0 = 0; k0 < K; k0 += 32) {
    v16h af = KG ? load_frag_g(arow, k0, kg, K) : load_frag(arow, k0, kg);
#pragma unroll
    for (int t = 0; t < 4; ++t) {
      v16h bf = KG ? load_frag_g(brow[t], k0, kg, K) : load_frag(brow[t], k0, kg);
      if (NG && !bval[t]) {
#pragma unroll
        for (int e = 0; e < 16; ++e) bf[e] = (half_t)0.f;
      }
      acc[t] = __builtin_amdgcn_wmma_f32_16x16x32_f16(
          false, af, false, bf, (short)0, acc[t], false, false);
    }
  }

#pragma unroll
  for (int t = 0; t < 4; ++t) {
    int n = n0 + t * 16 + lidx;
    if (NG && n >= N) continue;
#pragma unroll
    for (int v = 0; v < 8; ++v) {
      int m = mtile * 16 + v + kg * 8;       // D layout: m = vgpr + 8*(lane>=16)
      epi(m, n, acc[t][v]);
    }
  }
}

// ---------------- f32 -> f16 staging ----------------
__global__ void __launch_bounds__(256)
cvt_f16_kernel(const float* __restrict__ in, half_t* __restrict__ out, int n) {
  int i = blockIdx.x * 256 + threadIdx.x;
  if (i < n) out[i] = (half_t)in[i];
}

// ---------------- LayerNorm + affine ((x-mean)/(std+1e-10)) ----------------
__device__ __forceinline__ float block_sum256(float v, float* sm) {
  int t = threadIdx.x;
  sm[t] = v; __syncthreads();
  for (int s = 128; s > 0; s >>= 1) { if (t < s) sm[t] += sm[t + s]; __syncthreads(); }
  float r = sm[0]; __syncthreads();
  return r;
}

__global__ void __launch_bounds__(256)
ln_affine_kernel(const float* __restrict__ in, float* __restrict__ outf,
                 half_t* __restrict__ outh,
                 const float* __restrict__ w, const float* __restrict__ b,
                 const float* __restrict__ amean, const float* __restrict__ astd, int flip) {
  __shared__ float sm[256];
  int l = blockIdx.x;
  int src = flip ? (L - 1 - l) : l;
  int t = threadIdx.x;
  float x = in[(size_t)src * DM + t];
  float mean = block_sum256(x, sm) * (1.f / 256.f);
  float dx = x - mean;
  float var = block_sum256(dx * dx, sm) * (1.f / 256.f);
  float y = dx * rsqrtf(var + 1e-5f) * w[t] + b[t];
  y = (y - amean[t]) / (astd[t] + 1e-10f);
  if (outf) outf[(size_t)l * DM + t] = y;
  if (outh) outh[(size_t)l * DM + t] = (half_t)y;
}

// ---------------- depthwise causal conv (k=4) + SiLU, permuted domain ----------------
__global__ void __launch_bounds__(256)
conv_silu_kernel(const float* __restrict__ xp, const float* __restrict__ w,
                 const float* __restrict__ bias, float* __restrict__ xc) {
  int gid = blockIdx.x * 256 + threadIdx.x;          // L*DI threads
  int d = gid & (DI - 1);
  int p = gid >> 9;
  float acc = bias[d];
#pragma unroll
  for (int j = 0; j < 4; ++j) {
    int pp = p - 3 + j;
    if (pp >= 0) acc += w[d * 4 + j] * xp[(size_t)pp * DI + d];
  }
  xc[gid] = acc / (1.f + __expf(-acc));              // silu
}

// ---------------- chunked selective scan ----------------
// pass1: per (chunk,d): zero-state chunk response s[16] and sum(delta)
__global__ void __launch_bounds__(512)
scan_p1(const float* __restrict__ delta, const float* __restrict__ xc,
        const float* __restrict__ xdbl, const float* __restrict__ Alog,
        float* __restrict__ sumd, float* __restrict__ sacc) {
  __shared__ float sB[SC * 16];
  int c = blockIdx.x, d = threadIdx.x;
  for (int i = threadIdx.x; i < SC * 16; i += 512) {
    int r = i >> 4, n = i & 15;
    sB[i] = xdbl[(size_t)(c * SC + r) * 64 + 16 + n];
  }
  __syncthreads();
  float An[16], s[16];
#pragma unroll
  for (int n = 0; n < 16; ++n) { An[n] = -__expf(Alog[d * 16 + n]); s[n] = 0.f; }
  float sd = 0.f;
  for (int r = 0; r < SC; ++r) {
    int p = c * SC + r;
    float dl = delta[(size_t)p * DI + d];
    float xv = xc[(size_t)p * DI + d];
    float dx = dl * xv;
    sd += dl;
#pragma unroll
    for (int n = 0; n < 16; ++n) {
      float e = __expf(dl * An[n]);
      s[n] = e * s[n] + dx * sB[r * 16 + n];
    }
  }
  sumd[c * DI + d] = sd;
#pragma unroll
  for (int n = 0; n < 16; ++n) sacc[((size_t)c * DI + d) * 16 + n] = s[n];
}

// pass2: sequential combine over chunks, parallel over 8192 (d,n) states.
// Uses prod_t exp(dl*A) = exp(A * sum(dl)).
__global__ void __launch_bounds__(512)
scan_p2(const float* __restrict__ Alog, const float* __restrict__ sumd,
        const float* __restrict__ sacc, float* __restrict__ sinit) {
  int q = blockIdx.x * 512 + threadIdx.x;            // 8192
  int d = q >> 4, n = q & 15;
  float An = -__expf(Alog[d * 16 + n]);
  float cur = 0.f;
  for (int c = 0; c < CH; ++c) {
    float a = __expf(An * sumd[c * DI + d]);
    float sc = sacc[(size_t)c * 8192 + q];
    sinit[(size_t)c * 8192 + q] = cur;
    cur = a * cur + sc;
  }
}

// pass3: re-run chunks with correct init, fuse y = s.C + D*x, gate silu(z),
// un-permute, emit f16 directly for the out_proj WMMA A-operand.
__global__ void __launch_bounds__(512)
scan_p3(const float* __restrict__ delta, const float* __restrict__ xc,
        const float* __restrict__ xdbl, const float* __restrict__ Alog,
        const float* __restrict__ sinit, const float* __restrict__ Dp,
        const float* __restrict__ z, half_t* __restrict__ yout) {
  __shared__ float sBC[SC * 32];
  int c = blockIdx.x, d = threadIdx.x;
  for (int i = threadIdx.x; i < SC * 32; i += 512) {
    int r = i >> 5, n = i & 31;
    sBC[i] = xdbl[(size_t)(c * SC + r) * 64 + 16 + n];
  }
  __syncthreads();
  float An[16], s[16];
#pragma unroll
  for (int n = 0; n < 16; ++n) {
    An[n] = -__expf(Alog[d * 16 + n]);
    s[n] = sinit[(size_t)c * 8192 + d * 16 + n];
  }
  float Dv = Dp[d];
  for (int r = 0; r < SC; ++r) {
    int p = c * SC + r;
    float dl = delta[(size_t)p * DI + d];
    float xv = xc[(size_t)p * DI + d];
    float dx = dl * xv;
    float y = 0.f;
#pragma unroll
    for (int n = 0; n < 16; ++n) {
      float e = __expf(dl * An[n]);
      s[n] = e * s[n] + dx * sBC[r * 32 + n];
      y += s[n] * sBC[r * 32 + 16 + n];
    }
    float yv = y + Dv * xv;
    int l = 10 * (p % 1000) + p / 1000;              // idx[p]
    float zv = z[(size_t)l * DI + d];
    yout[(size_t)l * DI + d] = (half_t)(yv * (zv / (1.f + __expf(-zv))));
  }
}

// ---------------- attention head ----------------
__global__ void __launch_bounds__(64)
att_kernel(const float* __restrict__ h, const float* __restrict__ w1,
           const float* __restrict__ b1, const float* __restrict__ w2,
           const float* __restrict__ b2, float* __restrict__ att) {
  __shared__ float sm[64];
  int l = blockIdx.x, j = threadIdx.x;
  const float* row = h + (size_t)l * DM;
  float acc = b1[j];
  for (int d = 0; d < DM; ++d) acc += w1[j * DM + d] * row[d];
  sm[j] = tanhf(acc) * w2[j];
  __syncthreads();
  for (int s = 32; s > 0; s >>= 1) { if (j < s) sm[j] += sm[j + s]; __syncthreads(); }
  if (j == 0) att[l] = sm[0] + b2[0];
}

__global__ void __launch_bounds__(1024)
softmax_red_kernel(const float* __restrict__ att, float* __restrict__ red) {
  __shared__ float sm[1024];
  int t = threadIdx.x;
  float mx = -3.4e38f;
  for (int l = t; l < L; l += 1024) mx = fmaxf(mx, att[l]);
  sm[t] = mx; __syncthreads();
  for (int s = 512; s > 0; s >>= 1) { if (t < s) sm[t] = fmaxf(sm[t], sm[t + s]); __syncthreads(); }
  float m = sm[0]; __syncthreads();
  float sum = 0.f;
  for (int l = t; l < L; l += 1024) sum += __expf(att[l] - m);
  sm[t] = sum; __syncthreads();
  for (int s = 512; s > 0; s >>= 1) { if (t < s) sm[t] += sm[t + s]; __syncthreads(); }
  if (t == 0) { red[0] = m; red[1] = sm[0]; }
}

__global__ void __launch_bounds__(256)
feat_kernel(const float* __restrict__ att, const float* __restrict__ red,
            const float* __restrict__ h, float* __restrict__ outfeat,
            float* __restrict__ featbuf) {
  int d = threadIdx.x;
  float m = red[0], invZ = 1.f / red[1];
  float acc = 0.f;
  for (int l = 0; l < L; ++l) acc += __expf(att[l] - m) * h[(size_t)l * DM + d];
  float f = acc * invZ;
  outfeat[d] = f; featbuf[d] = f;
}

__global__ void logits_kernel(const float* __restrict__ feat, const float* __restrict__ cw,
                              const float* __restrict__ cb, float* __restrict__ out) {
  int c = threadIdx.x;                                // 2 threads
  float a = cb[c];
  for (int d = 0; d < DM; ++d) a += cw[c * DM + d] * feat[d];
  out[c] = a;
}

// ---------------- host launch ----------------
extern "C" void kernel_launch(void* const* d_in, const int* in_sizes, int n_in,
                              void* d_out, int out_size, void* d_ws, size_t ws_size,
                              hipStream_t stream) {
  (void)in_sizes; (void)n_in; (void)out_size; (void)ws_size;
  const float* x_path   = (const float*)d_in[0];
  const float* fc1_w    = (const float*)d_in[2];
  const float* fc1_b    = (const float*)d_in[3];
  const float* ln_w     = (const float*)d_in[4];
  const float* ln_b     = (const float*)d_in[5];
  const float* tta_mean = (const float*)d_in[6];
  const float* tta_std  = (const float*)d_in[7];
  const float* in_w     = (const float*)d_in[8];
  const float* conv_w   = (const float*)d_in[9];
  const float* conv_b   = (const float*)d_in[10];
  const float* xproj_w  = (const float*)d_in[11];
  const float* dtproj_w = (const float*)d_in[12];
  const float* dtproj_b = (const float*)d_in[13];
  const float* A_log    = (const float*)d_in[14];
  const float* D_param  = (const float*)d_in[15];
  const float* out_w    = (const float*)d_in[16];
  const float* norm_w   = (const float*)d_in[17];
  const float* norm_b   = (const float*)d_in[18];
  const float* aff_mean = (const float*)d_in[19];
  const float* aff_std  = (const float*)d_in[20];
  const float* att_w1   = (const float*)d_in[21];
  const float* att_b1   = (const float*)d_in[22];
  const float* att_w2   = (const float*)d_in[23];
  const float* att_b2   = (const float*)d_in[24];
  const float* cls_w    = (const float*)d_in[25];
  const float* cls_b    = (const float*)d_in[26];

  char* ws = (char*)d_ws;
  float*  H     = (float*)(ws + OF_H);
  float*  H0    = (float*)(ws + OF_H0);
  float*  XPERM = (float*)(ws + OF_XPERM);
  float*  DELTA = XPERM;                 // alias (xperm dead after conv)
  float*  Z     = (float*)(ws + OF_Z);
  float*  XC    = (float*)(ws + OF_XC);
  float*  XDBL  = (float*)(ws + OF_XDBL);
  half_t* TY16  = (half_t*)(ws + OF_TY16); // t16 (LN out), later y16 (scan out)
  float*  SUMD  = (float*)(ws + OF_SUMD);
  float*  SACC  = (float*)(ws + OF_SACC);
  float*  SINIT = (float*)(ws + OF_SINIT);
  float*  RED   = (float*)(ws + OF_RED);
  float*  FEAT  = (float*)(ws + OF_FEAT);
  float*  OUT   = (float*)d_out;         // [0,L): Att, [L,L+256): feat, +2 logits

  // f16 weight staging pointers
  half_t* WH      = (half_t*)(ws + OF_W16);
  half_t* wh_fc1  = WH;
  half_t* wh_in[2];  half_t* wh_xp[2];  half_t* wh_dt[2];  half_t* wh_out[2];
  for (int p = 0; p < 2; ++p) {
    half_t* base = WH + WN_FC1 + (size_t)p * WN_BLK;
    wh_in[p]  = base;
    wh_xp[p]  = base + WN_IN;
    wh_dt[p]  = base + WN_IN + WN_XP;
    wh_out[p] = base + WN_IN + WN_XP + WN_DT;
  }

  // ---- stage all weights to f16 (tiny, once per launch) ----
  {
    auto cvt = [&](const float* src, half_t* dst, size_t n) {
      cvt_f16_kernel<<<(unsigned)((n + 255) / 256), 256, 0, stream>>>(src, dst, (int)n);
    };
    cvt(fc1_w, wh_fc1, WN_FC1);
    for (int p = 0; p < 2; ++p) {
      const size_t i = (p == 0) ? 1 : 3;
      cvt(in_w     + i * WN_IN,  wh_in[p],  WN_IN);
      cvt(xproj_w  + i * WN_XP,  wh_xp[p],  WN_XP);
      cvt(dtproj_w + i * WN_DT,  wh_dt[p],  WN_DT);
      cvt(out_w    + i * WN_OUT, wh_out[p], WN_OUT);
    }
  }

  const int MG = 157;                    // ceil(625 m-tiles / 4 waves)

  // fc1 + GELU  (M=L, N=256, K=1024), A = f32 x_path
  gemm_wmma<float, false, false><<<dim3(MG, 4), 128, 0, stream>>>(
      x_path, DIN, wh_fc1, DIN, L, DM, DIN, EpiFC1{fc1_b, H});

  for (int pass = 0; pass < 2; ++pass) {
    const int i = (pass == 0) ? 1 : 3;
    const int flip = pass;               // block 3 consumes flipped h

    // LN + TTA affine -> t16 (f16 WMMA A-operand)
    ln_affine_kernel<<<L, 256, 0, stream>>>(
        H, nullptr, TY16, ln_w + i * DM, ln_b + i * DM,
        tta_mean + i * DM, tta_std + i * DM, flip);

    // in_proj (N=1024, K=256): scatter x into permuted order, z in natural order
    gemm_wmma<half_t, false, false><<<dim3(MG, 16), 128, 0, stream>>>(
        TY16, DM, wh_in[pass], DM, L, 2 * DI, DM, EpiInProj{XPERM, Z});

    conv_silu_kernel<<<(L * DI) / 256, 256, 0, stream>>>(
        XPERM, conv_w + (size_t)i * DI * 4, conv_b + (size_t)i * DI, XC);

    // xproj (N=48 padded to 64, K=512), A = f32 xc (read once)
    gemm_wmma<float, false, true><<<dim3(MG, 1), 128, 0, stream>>>(
        XC, DI, wh_xp[pass], DI, L, 48, DI, EpiXdbl{XDBL});

    // dtproj (N=512, K=16 guarded) + bias + softplus
    gemm_wmma<float, true, false><<<dim3(MG, 8), 128, 0, stream>>>(
        XDBL, 64, wh_dt[pass], 16, L, DI, 16,
        EpiDT{dtproj_b + (size_t)i * DI, DELTA});

    // chunked selective scan (3 passes), fused gate + un-permute, f16 y out
    scan_p1<<<CH, 512, 0, stream>>>(DELTA, XC, XDBL, A_log + (size_t)i * DI * 16, SUMD, SACC);
    scan_p2<<<16, 512, 0, stream>>>(A_log + (size_t)i * DI * 16, SUMD, SACC, SINIT);
    scan_p3<<<CH, 512, 0, stream>>>(DELTA, XC, XDBL, A_log + (size_t)i * DI * 16,
                                    SINIT, D_param + (size_t)i * DI, Z, TY16);

    // out_proj (N=256, K=512) + residual / bidirectional combine
    gemm_wmma<half_t, false, false><<<dim3(MG, 4), 128, 0, stream>>>(
        TY16, DI, wh_out[pass], DI, L, DM, DI, EpiOutB{H, H0, pass});
  }

  // final LN + affine -> H (f32, consumed by attention head)
  ln_affine_kernel<<<L, 256, 0, stream>>>(H0, H, nullptr, norm_w, norm_b,
                                          aff_mean, aff_std, 0);

  // attention logits (output #1), softmax stats, pooled feature (#2), logits (#3)
  att_kernel<<<L, 64, 0, stream>>>(H, att_w1, att_b1, att_w2, att_b2, OUT);
  softmax_red_kernel<<<1, 1024, 0, stream>>>(OUT, RED);
  feat_kernel<<<1, 256, 0, stream>>>(OUT, RED, H, OUT + L, FEAT);
  logits_kernel<<<1, 2, 0, stream>>>(FEAT, cls_w, cls_b, OUT + L + DM);
}
